// MDTA_85890755985955
// MI455X (gfx1250) — matrix operand order repaired and checked by reading, not compiled, not run on hardware
//
#include <hip/hip_runtime.h>
#include <math.h>

typedef __bf16 bf16;
typedef __attribute__((ext_vector_type(16))) __bf16 v16bf;
typedef __attribute__((ext_vector_type(8)))  float  v8f;

struct alignas(16) U4 { unsigned x, y, z, w; };
struct alignas(8)  U2 { unsigned x, y; };
struct alignas(16) F4 { float x, y, z, w; };

union Frag32 { v16bf v; U4 u4[2]; };
union H8     { U4 u4;  bf16 h[8]; };

#define EPS 1e-12f

__device__ __forceinline__ v8f wmma_bf16(v16bf a, v16bf b, v8f c) {
    // (neg_a, A, neg_b, B, c_mod, C, reuse_a, reuse_b)
    return __builtin_amdgcn_wmma_f32_16x16x32_bf16(false, a, false, b, (short)0, c,
                                                   false, false);
}

// CDNA5 async global->LDS copy (16B per active lane), tracked by ASYNCcnt.
__device__ __forceinline__ void async_copy_b128(void* lds_dst, const void* gsrc) {
    unsigned lds_off = (unsigned)(size_t)lds_dst;             // low 32 bits = LDS byte addr
    unsigned long long ga = (unsigned long long)(size_t)gsrc;
    asm volatile("global_load_async_to_lds_b128 %0, %1, off"
                 :: "v"(lds_off), "v"(ga) : "memory");
}
__device__ __forceinline__ void async_wait0() {
    asm volatile("s_wait_asynccnt 0x0" ::: "memory");
}

// ---------------------------------------------------------------- f32 -> bf16
__global__ __launch_bounds__(256)
void cvt_f32_bf16(const float* __restrict__ in, bf16* __restrict__ out, int n4) {
    int i = blockIdx.x * blockDim.x + threadIdx.x;
    if (i >= n4) return;
    F4 f = ((const F4*)in)[i];
    union { bf16 h[4]; U2 u; } o;
    o.h[0] = (bf16)f.x; o.h[1] = (bf16)f.y; o.h[2] = (bf16)f.z; o.h[3] = (bf16)f.w;
    ((U2*)out)[i] = o.u;
}

// ------------------------------------------------------- 1x1-conv GEMM (WMMA)
// Out[b, m, n] = sum_k W[m,k] * X[b,k,n], K=192, N=16384
// grid = (N/128, M/64, B), block = 256 (8 waves, 4 M x 2 N)
template <bool OUT_BF16>
__global__ __launch_bounds__(256)
void gemm_bf16_wmma(const bf16* __restrict__ W, const bf16* __restrict__ X,
                    void* __restrict__ OutV, int M) {
    constexpr int K = 192, N = 16384;
    __shared__ __align__(16) bf16 As[64 * 32];    // [m][k]
    __shared__ __align__(16) bf16 Bs[128 * 32];   // [n][k] (transposed)

    const int tid   = threadIdx.x;
    const int b     = blockIdx.z;
    const int m_blk = blockIdx.y * 64;
    const int n_blk = blockIdx.x * 128;
    const int wid   = tid >> 5, lane = tid & 31;
    const int wm = wid & 3, wn = wid >> 2;
    const int lr = lane & 15, hi = lane >> 4;

    const bf16* Xb = X + (size_t)b * K * N;

    v8f zero = {0.f, 0.f, 0.f, 0.f, 0.f, 0.f, 0.f, 0.f};
    v8f acc[4] = {zero, zero, zero, zero};

    const int a_row = tid >> 2, a_ko = (tid & 3) << 3;

    for (int k0 = 0; k0 < K; k0 += 32) {
        // stage A tile 64x32 via CDNA5 async global->LDS DMA (no transpose needed)
        async_copy_b128(As + a_row * 32 + a_ko,
                        W + (size_t)(m_blk + a_row) * K + k0 + a_ko);
        // stage B tile 32x128 transposed into [n][k]
        #pragma unroll
        for (int i = 0; i < 2; ++i) {
            int g = (tid << 1) + i;              // 0..511
            int k = g >> 4, n8 = (g & 15) << 3;
            H8 v; v.u4 = *(const U4*)(Xb + (size_t)(k0 + k) * N + n_blk + n8);
            if (k0 + 32 < K)
                __builtin_prefetch(Xb + (size_t)(k0 + 32 + k) * N + n_blk + n8, 0, 1);
            #pragma unroll
            for (int j = 0; j < 8; ++j) Bs[(n8 + j) * 32 + k] = v.h[j];
        }
        async_wait0();
        __syncthreads();

        Frag32 a;
        {
            const bf16* p = As + (wm * 16 + lr) * 32 + hi * 8;
            a.u4[0] = *(const U4*)p;
            a.u4[1] = *(const U4*)(p + 16);
        }
        #pragma unroll
        for (int j = 0; j < 4; ++j) {
            Frag32 bf;
            const bf16* p = Bs + (wn * 64 + j * 16 + lr) * 32 + hi * 8;
            bf.u4[0] = *(const U4*)p;
            bf.u4[1] = *(const U4*)(p + 16);
            acc[j] = wmma_bf16(a.v, bf.v, acc[j]);
        }
        __syncthreads();
    }

    const size_t outbase = (size_t)b * M * N;
    #pragma unroll
    for (int j = 0; j < 4; ++j) {
        int col = n_blk + wn * 64 + j * 16 + lr;
        #pragma unroll
        for (int r = 0; r < 8; ++r) {
            int row = m_blk + wm * 16 + r + 8 * hi;
            float v = acc[j][r];
            if (OUT_BF16) ((bf16*)OutV)[outbase + (size_t)row * N + col] = (bf16)v;
            else          ((float*)OutV)[outbase + (size_t)row * N + col] = v;
        }
    }
}

// --------------------------------------------- depthwise 3x3 + q/k sum-of-sq
// grid = (B*576, 8), block = 256: each block does 16 rows of one channel plane
__global__ __launch_bounds__(256)
void dwconv3x3(const bf16* __restrict__ in, const float* __restrict__ w9,
               bf16* __restrict__ out, float* __restrict__ sumsq) {
    const int plane = blockIdx.x;            // b*576 + c
    const int b = plane / 576, c = plane % 576;
    const int tid = threadIdx.x;
    const int row = blockIdx.y * 16 + (tid >> 4);
    const int cb  = (tid & 15) << 3;
    const size_t base = (size_t)plane * 16384;

    float w[9];
    #pragma unroll
    for (int i = 0; i < 9; ++i) w[i] = w9[c * 9 + i];

    float vals[3][10];
    #pragma unroll
    for (int dy = 0; dy < 3; ++dy) {
        int y = row + dy - 1;
        bool yok = (y >= 0) && (y < 128);
        #pragma unroll
        for (int dx = 0; dx < 10; ++dx) {
            int x = cb + dx - 1;
            vals[dy][dx] = (yok && x >= 0 && x < 128)
                               ? (float)in[base + (size_t)y * 128 + x] : 0.f;
        }
    }
    H8 o;
    float ss = 0.f;
    #pragma unroll
    for (int j = 0; j < 8; ++j) {
        float a = 0.f;
        #pragma unroll
        for (int dy = 0; dy < 3; ++dy)
            #pragma unroll
            for (int dx = 0; dx < 3; ++dx)
                a += w[dy * 3 + dx] * vals[dy][j + dx];
        o.h[j] = (bf16)a;
        ss += a * a;
    }
    *(U4*)(out + base + (size_t)row * 128 + cb) = o.u4;

    __shared__ float red[256];
    red[tid] = ss;
    __syncthreads();
    for (int s = 128; s > 0; s >>= 1) {
        if (tid < s) red[tid] += red[tid + s];
        __syncthreads();
    }
    if (tid == 0 && c < 384) {
        int idx = (c < 192) ? (b * 192 + c) : (1536 + b * 192 + (c - 192));
        unsafeAtomicAdd(&sumsq[idx], red[0]);
    }
}

// -------------------------------------------------- attention: S = q k^T
// grid = (8 K-slices, 32 bh), block = 256. Partial 48x48 in LDS, merged to Sg.
__global__ __launch_bounds__(256)
void attn_scores(const bf16* __restrict__ qkv, float* __restrict__ Sg) {
    __shared__ __align__(16) float S[48 * 48];
    const int tid = threadIdx.x;
    const int bh = blockIdx.y, b = bh >> 2, h = bh & 3;
    const int wid = tid >> 5, lane = tid & 31;
    const int lr = lane & 15, hi = lane >> 4;

    for (int i = tid; i < 48 * 48; i += 256) S[i] = 0.f;
    __syncthreads();

    const size_t qoff = ((size_t)b * 576 + h * 48) * 16384;
    const size_t koff = ((size_t)b * 576 + 192 + h * 48) * 16384;

    v8f zero = {0.f, 0.f, 0.f, 0.f, 0.f, 0.f, 0.f, 0.f};
    v8f acc[9] = {zero, zero, zero, zero, zero, zero, zero, zero, zero};

    const int n0 = blockIdx.x * 2048 + wid * 256;
    for (int it = 0; it < 8; ++it) {
        int n = n0 + it * 32;
        Frag32 af[3], bfm[3];
        #pragma unroll
        for (int ct = 0; ct < 3; ++ct) {
            const bf16* qp = qkv + qoff + (size_t)(ct * 16 + lr) * 16384 + n + hi * 8;
            af[ct].u4[0] = *(const U4*)qp;
            af[ct].u4[1] = *(const U4*)(qp + 16);
            const bf16* kp = qkv + koff + (size_t)(ct * 16 + lr) * 16384 + n + hi * 8;
            bfm[ct].u4[0] = *(const U4*)kp;
            bfm[ct].u4[1] = *(const U4*)(kp + 16);
        }
        #pragma unroll
        for (int ci = 0; ci < 3; ++ci)
            #pragma unroll
            for (int di = 0; di < 3; ++di)
                acc[ci * 3 + di] = wmma_bf16(af[ci].v, bfm[di].v, acc[ci * 3 + di]);
    }
    #pragma unroll
    for (int t = 0; t < 9; ++t) {
        int ct = t / 3, dt = t % 3;
        #pragma unroll
        for (int r = 0; r < 8; ++r) {
            int row = ct * 16 + r + 8 * hi;
            int col = dt * 16 + lr;
            __hip_atomic_fetch_add(&S[row * 48 + col], acc[t][r],
                                   __ATOMIC_RELAXED, __HIP_MEMORY_SCOPE_WORKGROUP);
        }
    }
    __syncthreads();
    float* dst = Sg + (size_t)bh * 2304;
    for (int i = tid; i < 2304; i += 256) unsafeAtomicAdd(&dst[i], S[i]);
}

// ------------------------------------ norms + softmax -> padded bf16 A-matrix
__global__ __launch_bounds__(64)
void attn_softmax(const float* __restrict__ Sg, const float* __restrict__ sumsq,
                  const float* __restrict__ temperature, bf16* __restrict__ attnG) {
    __shared__ float rnk[48];
    const int tid = threadIdx.x;
    const int bh = blockIdx.x, b = bh >> 2, h = bh & 3;
    if (tid < 48)
        rnk[tid] = 1.f / fmaxf(sqrtf(sumsq[1536 + b * 192 + h * 48 + tid]), EPS);
    __syncthreads();
    if (tid < 48) {
        const float* Srow = Sg + (size_t)bh * 2304 + tid * 48;
        float rq = (1.f / fmaxf(sqrtf(sumsq[b * 192 + h * 48 + tid]), EPS))
                   * temperature[h];
        float m = -1e30f;
        for (int d = 0; d < 48; ++d) m = fmaxf(m, Srow[d] * rq * rnk[d]);
        float sum = 0.f;
        for (int d = 0; d < 48; ++d) sum += __expf(Srow[d] * rq * rnk[d] - m);
        float rs = 1.f / sum;
        bf16* orow = attnG + (size_t)bh * 3072 + tid * 64;
        for (int d = 0; d < 48; ++d)
            orow[d] = (bf16)(__expf(Srow[d] * rq * rnk[d] - m) * rs);
        for (int d = 48; d < 64; ++d) orow[d] = (bf16)0.f;
    }
}

// ---------------------------------------------- out = attn @ v (WMMA)
// grid = (16 n-chunks of 1024, 32 bh), block = 256
__global__ __launch_bounds__(256)
void attn_apply(const bf16* __restrict__ qkv, const bf16* __restrict__ attnG,
                bf16* __restrict__ outp) {
    __shared__ __align__(16) bf16 attnA[48 * 64];
    __shared__ __align__(16) bf16 vT[128 * 64];     // [n][d], d>=48 stays zero

    const int tid = threadIdx.x;
    const int bh = blockIdx.y, b = bh >> 2, h = bh & 3;
    const int wid = tid >> 5, lane = tid & 31;
    const int lr = lane & 15, hi = lane >> 4;

    for (int i = tid; i < 384; i += 256)
        *(U4*)(attnA + i * 8) = *(const U4*)(attnG + (size_t)bh * 3072 + i * 8);
    for (int i = tid; i < 128 * 64 / 2; i += 256) ((unsigned*)vT)[i] = 0u;
    __syncthreads();

    Frag32 aT[3][2];
    #pragma unroll
    for (int ct = 0; ct < 3; ++ct)
        #pragma unroll
        for (int kq = 0; kq < 2; ++kq) {
            const bf16* p = attnA + (ct * 16 + lr) * 64 + kq * 32 + hi * 8;
            aT[ct][kq].u4[0] = *(const U4*)p;
            aT[ct][kq].u4[1] = *(const U4*)(p + 16);
        }

    const size_t voff  = ((size_t)b * 576 + 384 + h * 48) * 16384;
    const size_t obase = ((size_t)b * 192 + h * 48) * 16384;
    const int nbase = blockIdx.x * 1024;
    v8f zero = {0.f, 0.f, 0.f, 0.f, 0.f, 0.f, 0.f, 0.f};

    for (int nc = 0; nc < 8; ++nc) {
        int nb = nbase + nc * 128;
        #pragma unroll
        for (int i = 0; i < 3; ++i) {
            int g = tid * 3 + i;                 // 0..767
            int d = g >> 4, n8 = (g & 15) << 3;
            H8 v; v.u4 = *(const U4*)(qkv + voff + (size_t)d * 16384 + nb + n8);
            #pragma unroll
            for (int j = 0; j < 8; ++j) vT[(n8 + j) * 64 + d] = v.h[j];
        }
        __syncthreads();

        Frag32 bT[2];
        #pragma unroll
        for (int kq = 0; kq < 2; ++kq) {
            const bf16* p = vT + (wid * 16 + lr) * 64 + kq * 32 + hi * 8;
            bT[kq].u4[0] = *(const U4*)p;
            bT[kq].u4[1] = *(const U4*)(p + 16);
        }
        v8f oa[3] = {zero, zero, zero};
        #pragma unroll
        for (int ct = 0; ct < 3; ++ct)
            #pragma unroll
            for (int kq = 0; kq < 2; ++kq)
                oa[ct] = wmma_bf16(aT[ct][kq].v, bT[kq].v, oa[ct]);

        #pragma unroll
        for (int ct = 0; ct < 3; ++ct)
            #pragma unroll
            for (int r = 0; r < 8; ++r) {
                int row = ct * 16 + r + 8 * hi;
                int col = nb + wid * 16 + lr;
                outp[obase + (size_t)row * 16384 + col] = (bf16)oa[ct][r];
            }
        __syncthreads();
    }
}

// ---------------------------------------------------------------------------
extern "C" void kernel_launch(void* const* d_in, const int* in_sizes, int n_in,
                              void* d_out, int out_size, void* d_ws, size_t ws_size,
                              hipStream_t stream) {
    const float* x      = (const float*)d_in[0];
    const float* qkv_w  = (const float*)d_in[1];
    const float* dw_w   = (const float*)d_in[2];
    const float* proj_w = (const float*)d_in[3];
    const float* temp   = (const float*)d_in[4];

    char* ws = (char*)d_ws;
    size_t off = 0;
    auto carve = [&](size_t bytes) {
        void* p = ws + off;
        off = (off + bytes + 255) & ~(size_t)255;
        return p;
    };
    bf16*  x_bf     = (bf16*)carve((size_t)8 * 192 * 16384 * 2);
    bf16*  qkvw_bf  = (bf16*)carve((size_t)576 * 192 * 2);
    bf16*  projw_bf = (bf16*)carve((size_t)192 * 192 * 2);
    bf16*  qkv_lin  = (bf16*)carve((size_t)8 * 576 * 16384 * 2);
    bf16*  qkv_dw   = (bf16*)carve((size_t)8 * 576 * 16384 * 2);
    bf16*  attn_out = (bf16*)carve((size_t)8 * 192 * 16384 * 2);
    float* sumsq    = (float*)carve((size_t)2 * 8 * 192 * 4);
    float* Sg       = (float*)carve((size_t)32 * 2304 * 4);
    bf16*  attnG    = (bf16*)carve((size_t)32 * 3072 * 2);

    hipMemsetAsync(sumsq, 0, (size_t)2 * 8 * 192 * 4, stream);
    hipMemsetAsync(Sg, 0, (size_t)32 * 2304 * 4, stream);

    int n4 = 8 * 192 * 16384 / 4;
    cvt_f32_bf16<<<(n4 + 255) / 256, 256, 0, stream>>>(x, x_bf, n4);
    n4 = 576 * 192 / 4;
    cvt_f32_bf16<<<(n4 + 255) / 256, 256, 0, stream>>>(qkv_w, qkvw_bf, n4);
    n4 = 192 * 192 / 4;
    cvt_f32_bf16<<<(n4 + 255) / 256, 256, 0, stream>>>(proj_w, projw_bf, n4);

    gemm_bf16_wmma<true><<<dim3(128, 9, 8), 256, 0, stream>>>(qkvw_bf, x_bf,
                                                              (void*)qkv_lin, 576);
    dwconv3x3<<<dim3(8 * 576, 8), 256, 0, stream>>>(qkv_lin, dw_w, qkv_dw, sumsq);
    attn_scores<<<dim3(8, 32), 256, 0, stream>>>(qkv_dw, Sg);
    attn_softmax<<<32, 64, 0, stream>>>(Sg, sumsq, temp, attnG);
    attn_apply<<<dim3(16, 32), 256, 0, stream>>>(qkv_dw, attnG, attn_out);
    gemm_bf16_wmma<false><<<dim3(128, 3, 8), 256, 0, stream>>>(projw_bf, attn_out,
                                                               d_out, 192);
}
